// PointNet2Encoder_79688823210377
// MI455X (gfx1250) — compile-verified, matrix operand order checked by reading, not run on hardware
//
#include <hip/hip_runtime.h>

typedef __attribute__((ext_vector_type(16))) __bf16 v16bf;
typedef __attribute__((ext_vector_type(8)))  __bf16 v8bf;
typedef __attribute__((ext_vector_type(8)))  float  v8f;

#define BN_EPS 1e-5f
#define BATCH 4
#define NPTS  16384
#define KNBR  32

struct SwzLayer { const __bf16* W; const float* bias; };
struct SwzStage { SwzLayer l[3]; };

// ---------------- input transpose: (B,6,N) -> pts (B,N,6), xyz (B,N,3) -------
__global__ void transpose_in_kernel(const float* __restrict__ x,
                                    float* __restrict__ pts0,
                                    float* __restrict__ xyz0) {
  int idx = blockIdx.x * blockDim.x + threadIdx.x;
  if (idx >= BATCH * NPTS) return;
  int b = idx / NPTS, n = idx % NPTS;
  const float* src = x + (size_t)b * 6 * NPTS;
  float v[6];
#pragma unroll
  for (int c = 0; c < 6; ++c) v[c] = src[(size_t)c * NPTS + n];
  float* pd = pts0 + (size_t)idx * 6;
#pragma unroll
  for (int c = 0; c < 6; ++c) pd[c] = v[c];
  float* xd = xyz0 + (size_t)idx * 3;
  xd[0] = v[0]; xd[1] = v[1]; xd[2] = v[2];
}

// ---------------- weight pre-swizzle: BN-folded bf16 B-fragments -------------
// Layout: frag = nt*KC + kc ; per frag 32 lanes x 16 bf16 (contiguous 32B/lane)
__global__ void prep_layer_kernel(const float* __restrict__ W, const float* __restrict__ b,
                                  const float* __restrict__ g, const float* __restrict__ be,
                                  const float* __restrict__ mu, const float* __restrict__ vr,
                                  __bf16* __restrict__ Wswz, float* __restrict__ bias2,
                                  int Cin, int KPAD, int Cout) {
  int t = blockIdx.x * blockDim.x + threadIdx.x;
  if (t < Cout) {
    float sc = g[t] * rsqrtf(vr[t] + BN_EPS);
    bias2[t] = (b[t] - mu[t]) * sc + be[t];
  }
  int KC = KPAD / 32;
  int total = (Cout / 16) * KC * 32;
  if (t >= total) return;
  int lane = t & 31, frag = t >> 5;
  int kc = frag % KC, nt = frag / KC;
  int n = nt * 16 + (lane & 15);
  int half = lane >> 4;
  int k0 = kc * 32;
  float sc = g[n] * rsqrtf(vr[n] + BN_EPS);
  v16bf o{};
#pragma unroll
  for (int e = 0; e < 16; ++e) {
    int v = e >> 1, par = e & 1;
    int k = k0 + ((v >> 2) << 4) + (half << 3) + ((v & 3) << 1) + par;
    float w = (k < Cin) ? W[(size_t)k * Cout + n] * sc : 0.0f;
    o[e] = (__bf16)w;
  }
  *(v16bf*)(Wswz + (size_t)t * 16) = o;
}

// ---------------- farthest point sampling (1 block per batch) ----------------
__global__ __launch_bounds__(1024)
void fps_kernel(const float* __restrict__ xyz, int* __restrict__ fidx,
                float* __restrict__ dist, int Nin, int npoint) {
  const int T = 1024;
  int b = blockIdx.x, t = threadIdx.x;
  const float* X = xyz + (size_t)b * Nin * 3;
  float* D = dist + (size_t)b * Nin;
  int* F = fidx + (size_t)b * npoint;
  __shared__ float sv[1024];
  __shared__ int   si[1024];
  __shared__ float cpt[3];
  __shared__ int   sfar;
  for (int i = t; i < Nin; i += T) D[i] = 1e10f;
  if (t == 0) sfar = 0;
  __syncthreads();
  for (int it = 0; it < npoint; ++it) {
    int far = sfar;
    if (t == 0) {
      F[it] = far;
      cpt[0] = X[far*3]; cpt[1] = X[far*3+1]; cpt[2] = X[far*3+2];
    }
    __syncthreads();
    float cx = cpt[0], cy = cpt[1], cz = cpt[2];
    float best = -1.0f; int bi = 0;
    for (int i = t; i < Nin; i += T) {
      float dx = X[i*3] - cx, dy = X[i*3+1] - cy, dz = X[i*3+2] - cz;
      float d = dx*dx + dy*dy + dz*dz;
      float m = fminf(D[i], d);
      D[i] = m;
      if (m > best) { best = m; bi = i; }   // lowest index kept on ties
    }
    sv[t] = best; si[t] = bi;
    __syncthreads();
    for (int s = T/2; s > 0; s >>= 1) {
      if (t < s) {
        if (sv[t+s] > sv[t] || (sv[t+s] == sv[t] && si[t+s] < si[t])) {
          sv[t] = sv[t+s]; si[t] = si[t+s];
        }
      }
      __syncthreads();
    }
    if (t == 0) sfar = si[0];
    __syncthreads();
  }
}

// ---------------- gather sampled centers -------------------------------------
__global__ void gather_xyz_kernel(const float* __restrict__ xyz, const int* __restrict__ fidx,
                                  float* __restrict__ nxyz, int Nin, int npoint) {
  int idx = blockIdx.x * blockDim.x + threadIdx.x;
  if (idx >= BATCH * npoint) return;
  int b = idx / npoint;
  int p = fidx[idx];
  const float* s = xyz + ((size_t)b * Nin + p) * 3;
  float* d = nxyz + (size_t)idx * 3;
  d[0] = s[0]; d[1] = s[1]; d[2] = s[2];
}

// ---------------- ball query: one wave per center, wave32 ballot -------------
__global__ void ball_query_kernel(const float* __restrict__ xyz, const float* __restrict__ nxyz,
                                  int* __restrict__ gidx, int Nin, int npoint, float r2) {
  int wid = blockIdx.x * (blockDim.x >> 5) + (threadIdx.x >> 5);
  int lane = threadIdx.x & 31;
  if (wid >= BATCH * npoint) return;
  int b = wid / npoint;
  const float* X = xyz + (size_t)b * Nin * 3;
  const float* c = nxyz + (size_t)wid * 3;
  int* G = gidx + (size_t)wid * KNBR;
  float cx = c[0], cy = c[1], cz = c[2];
  int cnt = 0, first = -1;
  for (int base = 0; base < Nin && cnt < KNBR; base += 32) {
    int i = base + lane;
    bool ok = false;
    if (i < Nin) {
      float dx = X[i*3] - cx, dy = X[i*3+1] - cy, dz = X[i*3+2] - cz;
      ok = (dx*dx + dy*dy + dz*dz) <= r2;
    }
    unsigned mm = (unsigned)__ballot(ok);
    while (mm && cnt < KNBR) {
      int bit = __builtin_ctz(mm);
      mm &= mm - 1;
      int pidx = base + bit;
      if (first < 0) first = pidx;
      if (lane == 0) G[cnt] = pidx;
      ++cnt;
    }
  }
  if (first < 0) first = 0;
  if (lane == 0) for (; cnt < KNBR; ++cnt) G[cnt] = first;
}

// ---------------- WMMA A-fragment: two contiguous 16B LDS reads --------------
__device__ __forceinline__ v16bf load_a_lds(const __bf16* X, int stride, int m0, int k0, int lane) {
  int M = m0 + (lane & 15);
  int half = lane >> 4;
  const __bf16* p = X + M * stride + k0 + half * 8;
  v8bf lo = *(const v8bf*)p;
  v8bf hi = *(const v8bf*)(p + 16);
  return __builtin_shufflevector(lo, hi, 0,1,2,3,4,5,6,7,8,9,10,11,12,13,14,15);
}

// ---------------- one MLP layer on a 32-row group via WMMA -------------------
template<int KPAD>
__device__ __forceinline__ void mlp_layer(const __bf16* Xin, __bf16* Xout, const SwzLayer& L,
                                          int Cout, int lane) {
  constexpr int KC = KPAD / 32;
  int nloc = lane & 15, half = lane >> 4;
  __builtin_prefetch(L.W, 0, 1);
  for (int n0 = 0, nt = 0; n0 < Cout; n0 += 16, ++nt) {
    int n = n0 + nloc;
    float b2 = L.bias[n];
#pragma unroll
    for (int m0 = 0; m0 < 32; m0 += 16) {
      v8f acc = {};
#pragma unroll
      for (int kc = 0; kc < KC; ++kc) {
        v16bf a  = load_a_lds(Xin, KPAD, m0, kc * 32, lane);
        v16bf bf = *(const v16bf*)(L.W + ((size_t)(nt * KC + kc) * 32 + lane) * 16);
        acc = __builtin_amdgcn_wmma_f32_16x16x32_bf16(false, a, false, bf,
                                                      (short)0, acc, false, false);
      }
#pragma unroll
      for (int r = 0; r < 8; ++r) {
        float y = fmaxf(acc[r] + b2, 0.0f);
        int M = m0 + r + (half << 3);
        Xout[M * Cout + n] = (__bf16)y;
      }
    }
  }
}

// ---------------- final layer + max-over-K in registers ----------------------
template<int KPAD>
__device__ __forceinline__ void final_layer(const __bf16* Xin, const SwzLayer& L,
                                            int Cout, int lane, float* __restrict__ out) {
  constexpr int KC = KPAD / 32;
  int nloc = lane & 15, half = lane >> 4;
  __builtin_prefetch(L.W, 0, 1);
  for (int n0 = 0, nt = 0; n0 < Cout; n0 += 16, ++nt) {
    int n = n0 + nloc;
    float b2 = L.bias[n];
    float cmax = 0.0f;                       // post-ReLU values are >= 0
#pragma unroll
    for (int m0 = 0; m0 < 32; m0 += 16) {
      v8f acc = {};
#pragma unroll
      for (int kc = 0; kc < KC; ++kc) {
        v16bf a  = load_a_lds(Xin, KPAD, m0, kc * 32, lane);
        v16bf bf = *(const v16bf*)(L.W + ((size_t)(nt * KC + kc) * 32 + lane) * 16);
        acc = __builtin_amdgcn_wmma_f32_16x16x32_bf16(false, a, false, bf,
                                                      (short)0, acc, false, false);
      }
#pragma unroll
      for (int r = 0; r < 8; ++r) cmax = fmaxf(cmax, fmaxf(acc[r] + b2, 0.0f));
    }
    cmax = fmaxf(cmax, __shfl_xor(cmax, 16, 32));  // merge the two M-halves
    if (half == 0) out[n] = cmax;
  }
}

// ---------------- group -> 3-layer MLP -> max (one wave per group) -----------
template<int CIN, int C1, int C2, int C3>
__global__ __launch_bounds__(32)
void group_mlp_kernel(const float* __restrict__ xyz, const float* __restrict__ pts,
                      const float* __restrict__ nxyz, const int* __restrict__ gidx,
                      SwzStage S, int Nin, int npoint, float* __restrict__ outp) {
  constexpr int KPAD  = (CIN + 31) & ~31;
  constexpr int BUF_A = (KPAD > C2) ? KPAD : C2;
  __shared__ __bf16 bufA[32 * BUF_A];
  __shared__ __bf16 bufB[32 * C1];

  int wg = blockIdx.x;                  // b * npoint + s
  int b = wg / npoint;
  int lane = threadIdx.x;
  const int Cpts = CIN - 3;

  // Build X0 row (bf16): [rel_xyz(3), feats(Cpts)], zero-padded to KPAD
  int pi = gidx[(size_t)wg * KNBR + lane];
  const float* c = nxyz + (size_t)wg * 3;
  const float* px = xyz + ((size_t)b * Nin + pi) * 3;
  __bf16* row = bufA + lane * KPAD;
  row[0] = (__bf16)(px[0] - c[0]);
  row[1] = (__bf16)(px[1] - c[1]);
  row[2] = (__bf16)(px[2] - c[2]);
  const float* pf = pts + ((size_t)b * Nin + pi) * Cpts;
  for (int cc = 0; cc < Cpts; ++cc) row[3 + cc] = (__bf16)pf[cc];
  for (int cc = CIN; cc < KPAD; ++cc) row[cc] = (__bf16)0.0f;
  __syncthreads();

  mlp_layer<KPAD>(bufA, bufB, S.l[0], C1, lane);   // X0 -> Y1
  __syncthreads();
  mlp_layer<C1>(bufB, bufA, S.l[1], C2, lane);     // Y1 -> Y2
  __syncthreads();
  final_layer<C2>(bufA, S.l[2], C3, lane, outp + (size_t)wg * C3);
}

// ---------------- pack outputs ------------------------------------------------
__global__ void pack_out_kernel(const float* __restrict__ nxyz, const float* __restrict__ npts,
                                float* __restrict__ out) {
  int t = blockIdx.x * blockDim.x + threadIdx.x;
  const int n1 = BATCH * 3 * 16;
  const int n2 = BATCH * 512 * 16;
  if (t < n1) {
    int b = t / 48, rem = t % 48, cch = rem / 16, s = rem % 16;
    out[t] = nxyz[((size_t)b * 16 + s) * 3 + cch];
  } else if (t < n1 + n2) {
    int u = t - n1;
    int b = u / (512 * 16), rem = u % (512 * 16), cch = rem / 16, s = rem % 16;
    out[t] = npts[((size_t)b * 16 + s) * 512 + cch];
  }
}

// =============================================================================
extern "C" void kernel_launch(void* const* d_in, const int* in_sizes, int n_in,
                              void* d_out, int out_size, void* d_ws, size_t ws_size,
                              hipStream_t stream) {
  (void)ws_size; (void)n_in; (void)in_sizes; (void)out_size;
  const float* xin = (const float*)d_in[0];
  auto gp = [&](int s, int l, int f) { return (const float*)d_in[1 + (s*3 + l)*6 + f]; };

  // ---- layer dims
  const int CINs[4] = {9, 67, 131, 259};
  const int MLP[4][3] = {{32,32,64},{64,64,128},{128,128,256},{256,256,512}};
  auto pad32 = [](int c) { return (c + 31) & ~31; };

  // ---- workspace carve (64B aligned slots)
  size_t off = 0;
  auto alloc = [&](size_t bytes) -> void* {
    void* p = (char*)d_ws + off;
    off += (bytes + 63) & ~(size_t)63;
    return p;
  };
  float* pts0  = (float*)alloc((size_t)BATCH*NPTS*6*4);
  float* xyz0  = (float*)alloc((size_t)BATCH*NPTS*3*4);
  float* dist  = (float*)alloc((size_t)BATCH*NPTS*4);
  int*   fidx  = (int*)  alloc((size_t)BATCH*1024*4);
  int*   gidx  = (int*)  alloc((size_t)BATCH*1024*KNBR*4);
  float* nxyz1 = (float*)alloc((size_t)BATCH*1024*3*4);
  float* npts1 = (float*)alloc((size_t)BATCH*1024*64*4);
  float* nxyz2 = (float*)alloc((size_t)BATCH*256*3*4);
  float* npts2 = (float*)alloc((size_t)BATCH*256*128*4);
  float* nxyz3 = (float*)alloc((size_t)BATCH*64*3*4);
  float* npts3 = (float*)alloc((size_t)BATCH*64*256*4);
  float* nxyz4 = (float*)alloc((size_t)BATCH*16*3*4);
  float* npts4 = (float*)alloc((size_t)BATCH*16*512*4);

  SwzStage S[4];
  for (int s = 0; s < 4; ++s) {
    for (int l = 0; l < 3; ++l) {
      int Cin  = (l == 0) ? CINs[s] : MLP[s][l-1];
      int KPAD = pad32(Cin);
      int Cout = MLP[s][l];
      __bf16* Wswz = (__bf16*)alloc((size_t)KPAD * Cout * 2);
      float*  bias = (float*) alloc((size_t)Cout * 4);
      S[s].l[l] = SwzLayer{Wswz, bias};
      int total = (Cout / 16) * (KPAD / 32) * 32;
      int nthr = total > Cout ? total : Cout;
      prep_layer_kernel<<<(nthr + 255)/256, 256, 0, stream>>>(
          gp(s,l,0), gp(s,l,1), gp(s,l,2), gp(s,l,3), gp(s,l,4), gp(s,l,5),
          Wswz, bias, Cin, KPAD, Cout);
    }
  }

  transpose_in_kernel<<<(BATCH*NPTS + 255)/256, 256, 0, stream>>>(xin, pts0, xyz0);

  // ---- stage 1: N=16384 -> 1024, r=0.1, MLP 9->32->32->64
  fps_kernel<<<BATCH, 1024, 0, stream>>>(xyz0, fidx, dist, NPTS, 1024);
  gather_xyz_kernel<<<(BATCH*1024 + 255)/256, 256, 0, stream>>>(xyz0, fidx, nxyz1, NPTS, 1024);
  ball_query_kernel<<<(BATCH*1024 + 7)/8, 256, 0, stream>>>(xyz0, nxyz1, gidx, NPTS, 1024, 0.01f);
  group_mlp_kernel<9, 32, 32, 64><<<BATCH*1024, 32, 0, stream>>>(
      xyz0, pts0, nxyz1, gidx, S[0], NPTS, 1024, npts1);

  // ---- stage 2: 1024 -> 256, r=0.2, MLP 67->64->64->128
  fps_kernel<<<BATCH, 1024, 0, stream>>>(nxyz1, fidx, dist, 1024, 256);
  gather_xyz_kernel<<<(BATCH*256 + 255)/256, 256, 0, stream>>>(nxyz1, fidx, nxyz2, 1024, 256);
  ball_query_kernel<<<(BATCH*256 + 7)/8, 256, 0, stream>>>(nxyz1, nxyz2, gidx, 1024, 256, 0.04f);
  group_mlp_kernel<67, 64, 64, 128><<<BATCH*256, 32, 0, stream>>>(
      nxyz1, npts1, nxyz2, gidx, S[1], 1024, 256, npts2);

  // ---- stage 3: 256 -> 64, r=0.4, MLP 131->128->128->256
  fps_kernel<<<BATCH, 1024, 0, stream>>>(nxyz2, fidx, dist, 256, 64);
  gather_xyz_kernel<<<(BATCH*64 + 255)/256, 256, 0, stream>>>(nxyz2, fidx, nxyz3, 256, 64);
  ball_query_kernel<<<(BATCH*64 + 7)/8, 256, 0, stream>>>(nxyz2, nxyz3, gidx, 256, 64, 0.16f);
  group_mlp_kernel<131, 128, 128, 256><<<BATCH*64, 32, 0, stream>>>(
      nxyz2, npts2, nxyz3, gidx, S[2], 256, 64, npts3);

  // ---- stage 4: 64 -> 16, r=0.8, MLP 259->256->256->512
  fps_kernel<<<BATCH, 1024, 0, stream>>>(nxyz3, fidx, dist, 64, 16);
  gather_xyz_kernel<<<(BATCH*16 + 255)/256, 256, 0, stream>>>(nxyz3, fidx, nxyz4, 64, 16);
  ball_query_kernel<<<(BATCH*16 + 7)/8, 256, 0, stream>>>(nxyz3, nxyz4, gidx, 64, 16, 0.64f);
  group_mlp_kernel<259, 256, 256, 512><<<BATCH*16, 32, 0, stream>>>(
      nxyz3, npts3, nxyz4, gidx, S[3], 64, 16, npts4);

  pack_out_kernel<<<(BATCH*(3*16 + 512*16) + 255)/256, 256, 0, stream>>>(nxyz4, npts4, (float*)d_out);
}
